// MoEsparseRouting_81578608820387
// MI455X (gfx1250) — matrix operand order repaired and checked by reading, not compiled, not run on hardware
//
#include <hip/hip_runtime.h>
#include <hip/hip_bf16.h>

typedef _Float16 v8h  __attribute__((ext_vector_type(8)));
typedef _Float16 v16h __attribute__((ext_vector_type(16)));
typedef float    v8f  __attribute__((ext_vector_type(8)));

#define B_    32
#define S_    512
#define M_    768
#define E_    4
#define ALPHA_ 16.0f

// ---------------------------------------------------------------- fp32 -> fp16
__global__ void k_cvt_f16(const float* __restrict__ in, _Float16* __restrict__ out, int n) {
  int i = blockIdx.x * blockDim.x + threadIdx.x;
  int stride = gridDim.x * blockDim.x;
  for (; i < n; i += stride) out[i] = (_Float16)in[i];
}

// ---------------------------------------------------------------- mean over S
__global__ void k_pool(const float* __restrict__ X, float* __restrict__ pooled) {
  int b = blockIdx.x;
  int m = blockIdx.y * 256 + threadIdx.x;        // gridDim.y*256 == M_
  const float* xb = X + (size_t)b * S_ * M_ + m;
  float s = 0.f;
  for (int t = 0; t < S_; ++t) s += xb[(size_t)t * M_];
  pooled[b * M_ + m] = s * (1.0f / (float)S_);
}

// ------------------------------------------- router logits + hard (argmax) gate
__global__ void k_router(const float* __restrict__ pooled,
                         const float* __restrict__ rw,
                         const float* __restrict__ rb,
                         int* __restrict__ eidx) {
  __shared__ float red[256][E_];
  int b = blockIdx.x, t = threadIdx.x;
  float acc[E_] = {0.f, 0.f, 0.f, 0.f};
  for (int m = t; m < M_; m += 256) {
    float p = pooled[b * M_ + m];
#pragma unroll
    for (int e = 0; e < E_; ++e) acc[e] += p * rw[e * M_ + m];
  }
#pragma unroll
  for (int e = 0; e < E_; ++e) red[t][e] = acc[e];
  __syncthreads();
  for (int off = 128; off > 0; off >>= 1) {
    if (t < off)
#pragma unroll
      for (int e = 0; e < E_; ++e) red[t][e] += red[t + off][e];
    __syncthreads();
  }
  if (t == 0) {
    int best = 0; float bv = red[0][0] + rb[0];
    for (int e = 1; e < E_; ++e) {
      float v = red[0][e] + rb[e];
      if (v > bv) { bv = v; best = e; }
    }
    eidx[b] = best;
  }
}

// ---------------------------------------------------------- WMMA GEMM: C = A*W^T + bias
// A: [16384 x 768] f16 row-major, W: [768 x 768] f16 row-major (row n = output col n).
// 128 threads = 4 waves in a 2(M) x 2(N) grid; wave tile 32x64 (8 accumulators);
// block tile 64x128. Double-buffered fragments: loads for step k+1 are issued
// before the WMMAs of step k, so waits are partial instead of wait-to-zero.
__global__ void __launch_bounds__(128)
k_gemm_wmma(const _Float16* __restrict__ A, const _Float16* __restrict__ W,
            const float* __restrict__ bias, float* __restrict__ C) {
  const int lane  = threadIdx.x & 31;
  const int wave  = threadIdx.x >> 5;
  const int waveM = wave & 1, waveN = wave >> 1;      // 2 x 2 wave grid
  const int mBase = blockIdx.x * 64 + waveM * 32;
  const int nBase = blockIdx.y * 128 + waveN * 64;
  const int lr    = lane & 15;
  const bool hi   = lane >= 16;

  // A fragment (16x32 f16, ISA 7.12.2): lane carries row M=lane&15;
  // lanes 0-15: K [k..k+15]; lanes 16-31: K [k+16..k+31] (contiguous, matches B)
  const _Float16* __restrict__ arow0 = A + (size_t)(mBase + lr) * M_;
  const _Float16* __restrict__ arow1 = arow0 + (size_t)16 * M_;
  const int ka = hi ? 16 : 0;
  // B fragment (32x16): lane holds column N=lr, 16 contiguous K values
  // = 16 contiguous f16 of row (nBase+t*16+lr) of W (row-major W gives W^T cols)
  const int kb = hi ? 16 : 0;
  const _Float16* __restrict__ brow0 = W + (size_t)(nBase + lr) * M_ + kb;

  v8f acc[8];
#pragma unroll
  for (int i = 0; i < 8; ++i) acc[i] = (v8f){};

  // ---- prologue: load k=0 fragments
  v8h a0lo = *(const v8h*)(arow0 + ka);
  v8h a0hi = *(const v8h*)(arow0 + ka + 8);
  v8h a1lo = *(const v8h*)(arow1 + ka);
  v8h a1hi = *(const v8h*)(arow1 + ka + 8);
  v16h a0c = __builtin_shufflevector(a0lo, a0hi, 0,1,2,3,4,5,6,7,8,9,10,11,12,13,14,15);
  v16h a1c = __builtin_shufflevector(a1lo, a1hi, 0,1,2,3,4,5,6,7,8,9,10,11,12,13,14,15);
  v16h bc[4];
#pragma unroll
  for (int tN = 0; tN < 4; ++tN)
    bc[tN] = *(const v16h*)(brow0 + (size_t)tN * 16 * M_);

#pragma unroll
  for (int k = 0; k < M_; k += 32) {
    v16h a0n, a1n, bn[4];
    if (k + 32 < M_) {
      // ---- prefetch next k-step fragments BEFORE consuming current ones
      const int kn = k + 32;
      v8h p0lo = *(const v8h*)(arow0 + kn + ka);
      v8h p0hi = *(const v8h*)(arow0 + kn + ka + 8);
      v8h p1lo = *(const v8h*)(arow1 + kn + ka);
      v8h p1hi = *(const v8h*)(arow1 + kn + ka + 8);
      a0n = __builtin_shufflevector(p0lo, p0hi, 0,1,2,3,4,5,6,7,8,9,10,11,12,13,14,15);
      a1n = __builtin_shufflevector(p1lo, p1hi, 0,1,2,3,4,5,6,7,8,9,10,11,12,13,14,15);
#pragma unroll
      for (int tN = 0; tN < 4; ++tN)
        bn[tN] = *(const v16h*)(brow0 + (size_t)tN * 16 * M_ + kn);
    }
    // ---- 8 WMMAs on current fragments (B reused by both A row-groups)
#pragma unroll
    for (int tN = 0; tN < 4; ++tN) {
      acc[tN] = __builtin_amdgcn_wmma_f32_16x16x32_f16(
          false, a0c, false, bc[tN], (short)0, acc[tN], false, false);
      acc[4 + tN] = __builtin_amdgcn_wmma_f32_16x16x32_f16(
          false, a1c, false, bc[tN], (short)0, acc[4 + tN], false, false);
    }
    a0c = a0n; a1c = a1n;
#pragma unroll
    for (int tN = 0; tN < 4; ++tN) bc[tN] = bn[tN];
  }

  // C/D layout: lane = col (lr); VGPR v -> row base + v + (hi ? 8 : 0)
#pragma unroll
  for (int tN = 0; tN < 4; ++tN) {
    const int col = nBase + tN * 16 + lr;
    const float bb = bias[col];
    const int r0 = mBase + (hi ? 8 : 0);
#pragma unroll
    for (int v = 0; v < 8; ++v) {
      C[(size_t)(r0 + v) * M_ + col]      = acc[tN][v] + bb;
      C[(size_t)(r0 + 16 + v) * M_ + col] = acc[4 + tN][v] + bb;
    }
  }
}

// ------------------------------------------------- TT-MoE chain + fused output
// gates are one-hot => just contract through the selected expert's 6 cores.
#define TT_T 8
__global__ void __launch_bounds__(256)
k_tt(const float* __restrict__ X, const int* __restrict__ eidx,
     const float* __restrict__ c0, const float* __restrict__ c1,
     const float* __restrict__ c2, const float* __restrict__ c3,
     const float* __restrict__ c4, const float* __restrict__ c5,
     float* __restrict__ out) {
  __shared__ float sA[TT_T][M_];
  __shared__ float sB[TT_T][M_];
  __shared__ float s0[96], s1[512], s2[512], s3[512], s4[512], s5[96];

  const int b = blockIdx.x;
  const int sBase = blockIdx.y * TT_T;
  const int tid = threadIdx.x;
  const int tok = tid >> 5;        // 8 tokens per block
  const int j = tid & 31;          // 32 lanes per token
  const int e = eidx[b];

  for (int i = tid; i < 96; i += 256) { s0[i] = c0[e * 96 + i]; s5[i] = c5[e * 96 + i]; }
  for (int i = tid; i < 512; i += 256) {
    s1[i] = c1[e * 512 + i]; s2[i] = c2[e * 512 + i];
    s3[i] = c3[e * 512 + i]; s4[i] = c4[e * 512 + i];
  }
  const size_t rowBase = ((size_t)b * S_ + sBase + tok) * M_;
  for (int i = j; i < M_; i += 32) sA[tok][i] = X[rowBase + i];
  __syncthreads();

  // view X row as [f3=8, f2=8, f1=12], f1 innermost: flat = f3*96 + f2*12 + f1
  // stage 1: out[p,f3,f2] = sum_m st[f3,f2,m] * c0[m,p]        (512 outputs)
  for (int o = j; o < 512; o += 32) {
    int p = o >> 6, f3 = (o >> 3) & 7, f2 = o & 7;
    const float* st = &sA[tok][f3 * 96 + f2 * 12];
    float s = 0.f;
    for (int m = 0; m < 12; ++m) s += st[m] * s0[m * 8 + p];
    sB[tok][o] = s;
  }
  __syncthreads();
  // stage 2: out[p,f3] = sum_{r,m} st[r,f3,m] * c1[r,m,p]      (64 outputs)
  for (int o = j; o < 64; o += 32) {
    int p = o >> 3, f3 = o & 7;
    float s = 0.f;
    for (int r = 0; r < 8; ++r) {
      const float* st = &sB[tok][r * 64 + f3 * 8];
      const float* cc = &s1[r * 64 + p];
      for (int m = 0; m < 8; ++m) s += st[m] * cc[m * 8];
    }
    sA[tok][o] = s;
  }
  __syncthreads();
  // stage 3: out[p] = sum_{r,m} st[r,m] * c2[r,m,p]            (8 outputs)
  if (j < 8) {
    float s = 0.f;
    for (int r = 0; r < 8; ++r)
      for (int m = 0; m < 8; ++m) s += sA[tok][r * 8 + m] * s2[r * 64 + m * 8 + j];
    sB[tok][j] = s;
  }
  __syncthreads();
  // stage 4: out[p,n] = sum_r st[r] * c3[r,n,p]                (64 outputs)
  for (int o = j; o < 64; o += 32) {
    int p = o >> 3, n = o & 7;
    float s = 0.f;
    for (int r = 0; r < 8; ++r) s += sB[tok][r] * s3[r * 64 + n * 8 + p];
    sA[tok][o] = s;
  }
  __syncthreads();
  // stage 5: out[p,n1,n] = sum_r st[r,n1] * c4[r,n,p]          (512 outputs)
  for (int o = j; o < 512; o += 32) {
    int p = o >> 6, n1 = (o >> 3) & 7, n = o & 7;
    float s = 0.f;
    for (int r = 0; r < 8; ++r) s += sA[tok][r * 8 + n1] * s4[r * 64 + n * 8 + p];
    sB[tok][o] = s;
  }
  __syncthreads();
  // stage 6: z[n1,n2,n] = sum_r st[r,n1,n2] * c5[r,n]; out = base + 16*z (768)
  for (int o = j; o < M_; o += 32) {
    int n1 = o / 96, rem = o % 96, n2 = rem / 12, n = rem % 12;
    float s = 0.f;
    for (int r = 0; r < 8; ++r) s += sB[tok][r * 64 + n1 * 8 + n2] * s5[r * 12 + n];
    out[rowBase + o] = out[rowBase + o] + ALPHA_ * s;
  }
}

// ---------------------------------------------------------------------- launch
extern "C" void kernel_launch(void* const* d_in, const int* in_sizes, int n_in,
                              void* d_out, int out_size, void* d_ws, size_t ws_size,
                              hipStream_t stream) {
  (void)in_sizes; (void)n_in; (void)out_size; (void)ws_size;
  const float* X        = (const float*)d_in[0];
  const float* router_w = (const float*)d_in[1];
  const float* router_b = (const float*)d_in[2];
  const float* base_w   = (const float*)d_in[3];
  const float* base_b   = (const float*)d_in[4];
  const float* core0    = (const float*)d_in[5];
  const float* core1    = (const float*)d_in[6];
  const float* core2    = (const float*)d_in[7];
  const float* core3    = (const float*)d_in[8];
  const float* core4    = (const float*)d_in[9];
  const float* core5    = (const float*)d_in[10];
  float* out = (float*)d_out;

  // workspace layout (all offsets 256B-aligned)
  const size_t nX = (size_t)B_ * S_ * M_;          // 12,582,912
  const size_t nW = (size_t)M_ * M_;               // 589,824
  char* ws = (char*)d_ws;
  _Float16* Xh     = (_Float16*)(ws);                          // 24 MB
  _Float16* Wh     = (_Float16*)(ws + nX * sizeof(_Float16));  // 1.125 MB
  float*    pooled = (float*)(ws + nX * sizeof(_Float16) + nW * sizeof(_Float16));
  int*      eidx   = (int*)((char*)pooled + (size_t)B_ * M_ * sizeof(float));

  k_cvt_f16<<<4096, 256, 0, stream>>>(X, Xh, (int)nX);
  k_cvt_f16<<<1024, 256, 0, stream>>>(base_w, Wh, (int)nW);

  k_pool<<<dim3(B_, M_ / 256), 256, 0, stream>>>(X, pooled);
  k_router<<<B_, 256, 0, stream>>>(pooled, router_w, router_b, eidx);

  // C tiles: 16384/64 = 256 along M, 768/128 = 6 along N
  k_gemm_wmma<<<dim3(256, 6), 128, 0, stream>>>(Xh, Wh, base_b, out);

  k_tt<<<dim3(B_, S_ / TT_T), 256, 0, stream>>>(X, eidx, core0, core1, core2,
                                                core3, core4, core5, out);
}
